// RMoE_78185584656651
// MI455X (gfx1250) — matrix operand order repaired and checked by reading, not compiled, not run on hardware
//
#include <hip/hip_runtime.h>
#include <cstdint>
#include <cstddef>

// ---------------------------------------------------------------------------
// Types for CDNA5 WMMA (wave32): v_wmma_f32_16x16x32_bf16
// ---------------------------------------------------------------------------
typedef __bf16 bf16x16 __attribute__((ext_vector_type(16)));
typedef __bf16 bf16x2  __attribute__((ext_vector_type(2)));
typedef float  f32x8   __attribute__((ext_vector_type(8)));
typedef float  f32x2   __attribute__((ext_vector_type(2)));

// Pairwise fptrunc -> single v_cvt_pk_bf16_f32 (both operands), no int glue.
__device__ __forceinline__ uint32_t pack2bf(float a, float b) {
    f32x2 f; f.x = a; f.y = b;
    bf16x2 h = __builtin_convertvector(f, bf16x2);
    return __builtin_bit_cast(uint32_t, h);
}

__device__ __forceinline__ unsigned short f2bf(float f) {
    __bf16 h = (__bf16)f;
    return __builtin_bit_cast(unsigned short, h);
}

// ---------------------------------------------------------------------------
// Generic batched WMMA GEMM:  C = alpha * A(MxK) * B(KxN) [+ bias][+ resid][relu]
// Block tile 128x64, 128 threads (4 waves); wave owns 32 rows ->
//   2 A-fragments x 4 B-fragments = 8 v_wmma per 32-deep K-step.
// Software pipeline: stage tile k0 to LDS, prefetch tile k0+32 into registers
// (float4 b128 loads) while the WMMAs consume LDS.
// Requirements (all call sites satisfy): M%128==0, N%64==0, K%32==0,
// 16B alignment of A/B rows (lda/ldb multiples of 4, bases multiples of 4).
// Batch offset for z = blockIdx.z: ptr += (z/div)*s1 + (z%div)*s2
// transB: B element (k,n) read as B[n*ldb + k] (B stored NxK row-major).
// ---------------------------------------------------------------------------
__global__ __launch_bounds__(128)
void wmma_gemm_kernel(const float* __restrict__ A, int lda, long long sA1, long long sA2,
                      const float* __restrict__ B, int ldb, long long sB1, long long sB2, int transB,
                      float* __restrict__ C, int ldc, long long sC1, long long sC2,
                      const float* __restrict__ bias,
                      const float* __restrict__ resid, int ldr, long long sR1, long long sR2,
                      int M, int N, int K, int batchDiv, float alpha, int relu)
{
    (void)M; (void)N;
    const int z  = blockIdx.z;
    const int zq = z / batchDiv, zr = z % batchDiv;
    A += zq * sA1 + zr * sA2;
    B += zq * sB1 + zr * sB2;
    C += zq * sC1 + zr * sC2;
    if (resid) resid += zq * sR1 + zr * sR2;

    const int m0 = blockIdx.y * 128;
    const int n0 = blockIdx.x * 64;

    constexpr int LD = 40;   // padded ushort stride: 80B rows, 16B aligned
    __shared__ __align__(16) unsigned short As[128 * LD];  // A tile [m][k]
    __shared__ __align__(16) unsigned short Bs[64 * LD];   // B tile [n][k]

    const int tid  = threadIdx.x;
    const int wave = tid >> 5;
    const int lane = tid & 31;
    const int half = lane >> 4;
    const int l15  = lane & 15;

    f32x8 acc[2][4];
#pragma unroll
    for (int m = 0; m < 2; ++m)
#pragma unroll
        for (int j = 0; j < 4; ++j)
#pragma unroll
            for (int e = 0; e < 8; ++e) acc[m][j][e] = 0.0f;

    float4 aReg[8];   // 128x32 tile: 1024 float4, 8 per thread
    float4 bReg[4];   // 64x32 (or 32x64) tile: 512 float4, 4 per thread

    // ---- tile prefetch (global -> registers, b128 loads) ----
    auto loadA = [&](int k0) {
#pragma unroll
        for (int i = 0; i < 8; ++i) {
            const int e = tid + i * 128;        // 0..1023
            const int r = e >> 3, c4 = e & 7;   // row, col/4
            aReg[i] = *(const float4*)(A + (long long)(m0 + r) * lda + k0 + c4 * 4);
        }
    };
    auto loadB = [&](int k0) {
        if (transB) {
#pragma unroll
            for (int i = 0; i < 4; ++i) {
                const int e = tid + i * 128;      // 0..511
                const int n = e >> 3, c4 = e & 7;
                bReg[i] = *(const float4*)(B + (long long)(n0 + n) * ldb + k0 + c4 * 4);
            }
        } else {
#pragma unroll
            for (int i = 0; i < 4; ++i) {
                const int e = tid + i * 128;
                const int k = e >> 4, c4 = e & 15;
                bReg[i] = *(const float4*)(B + (long long)(k0 + k) * ldb + n0 + c4 * 4);
            }
        }
    };
    // ---- registers -> LDS (bf16, packed 8B stores where possible) ----
    auto storeTiles = [&]() {
#pragma unroll
        for (int i = 0; i < 8; ++i) {
            const int e = tid + i * 128;
            const int r = e >> 3, c4 = e & 7;
            uint2 t; t.x = pack2bf(aReg[i].x, aReg[i].y); t.y = pack2bf(aReg[i].z, aReg[i].w);
            *(uint2*)&As[r * LD + c4 * 4] = t;
        }
        if (transB) {
#pragma unroll
            for (int i = 0; i < 4; ++i) {
                const int e = tid + i * 128;
                const int n = e >> 3, c4 = e & 7;
                uint2 t; t.x = pack2bf(bReg[i].x, bReg[i].y); t.y = pack2bf(bReg[i].z, bReg[i].w);
                *(uint2*)&Bs[n * LD + c4 * 4] = t;
            }
        } else {
#pragma unroll
            for (int i = 0; i < 4; ++i) {
                const int e = tid + i * 128;
                const int k = e >> 4, c4 = e & 15;
                Bs[(c4 * 4 + 0) * LD + k] = f2bf(bReg[i].x);
                Bs[(c4 * 4 + 1) * LD + k] = f2bf(bReg[i].y);
                Bs[(c4 * 4 + 2) * LD + k] = f2bf(bReg[i].z);
                Bs[(c4 * 4 + 3) * LD + k] = f2bf(bReg[i].w);
            }
        }
    };

    loadA(0);
    loadB(0);

    for (int k0 = 0; k0 < K; k0 += 32) {
        storeTiles();
        __syncthreads();
        if (k0 + 32 < K) {            // prefetch next K-tile while WMMAs run
            loadA(k0 + 32);
            loadB(k0 + 32);
        }

        // A fragments: lane (l15,half) row = base+l15 ; k = i + 8*half (+16)
        union { bf16x16 v; uint32_t u[8]; } a0, a1;
        {
            const unsigned short* ap0 = &As[(wave * 32 + l15) * LD];
            const unsigned short* ap1 = &As[(wave * 32 + 16 + l15) * LD];
            const uint32_t* p00 = (const uint32_t*)(ap0 + 8 * half);
            const uint32_t* p01 = (const uint32_t*)(ap0 + 16 + 8 * half);
            const uint32_t* p10 = (const uint32_t*)(ap1 + 8 * half);
            const uint32_t* p11 = (const uint32_t*)(ap1 + 16 + 8 * half);
#pragma unroll
            for (int i = 0; i < 4; ++i) {
                a0.u[i] = p00[i]; a0.u[4 + i] = p01[i];
                a1.u[i] = p10[i]; a1.u[4 + i] = p11[i];
            }
        }
#pragma unroll
        for (int j = 0; j < 4; ++j) {
            // B fragment: lane col n = j*16+l15 ; k = i + 8*half (+16)
            union { bf16x16 v; uint32_t u[8]; } bf;
            const uint32_t* q0 = (const uint32_t*)(&Bs[(j * 16 + l15) * LD + 8 * half]);
            const uint32_t* q1 = (const uint32_t*)(&Bs[(j * 16 + l15) * LD + 16 + 8 * half]);
#pragma unroll
            for (int i = 0; i < 4; ++i) { bf.u[i] = q0[i]; bf.u[4 + i] = q1[i]; }

            acc[0][j] = __builtin_amdgcn_wmma_f32_16x16x32_bf16(
                false, a0.v, false, bf.v, (short)0, acc[0][j], false, false);
            acc[1][j] = __builtin_amdgcn_wmma_f32_16x16x32_bf16(
                false, a1.v, false, bf.v, (short)0, acc[1][j], false, false);
        }
        __syncthreads();
    }

    // --- epilogue: C/D layout row = p + 8*half, col = l15 ---
#pragma unroll
    for (int m = 0; m < 2; ++m) {
#pragma unroll
        for (int j = 0; j < 4; ++j) {
            const int col = n0 + j * 16 + l15;
            const float bv = bias ? bias[col] : 0.0f;
#pragma unroll
            for (int p = 0; p < 8; ++p) {
                const int row = m0 + wave * 32 + m * 16 + p + 8 * half;
                float v = acc[m][j][p] * alpha + bv;
                if (resid) v += resid[(long long)row * ldr + col];
                if (relu)  v = fmaxf(v, 0.0f);
                C[(long long)row * ldc + col] = v;
            }
        }
    }
}

// ---------------------------------------------------------------------------
// Embedding + sinusoidal positional encoding:  h = emb[x]*sqrt(D) + PE
// ---------------------------------------------------------------------------
__global__ __launch_bounds__(256)
void embed_kernel(const int* __restrict__ x, const float* __restrict__ emb,
                  float* __restrict__ H, int S, int D)
{
    const int t   = blockIdx.x;            // b*S + s
    const int s   = t % S;
    const int tok = x[t];
    for (int d = threadIdx.x; d < D; d += 256) {
        float ang = (float)s / powf(10000.0f, (2.0f * (float)(d >> 1)) / (float)D);
        float pe  = (d & 1) ? cosf(ang) : sinf(ang);
        H[(long long)t * D + d] = emb[(long long)tok * D + d] * 32.0f + pe; // sqrt(1024)=32
    }
}

// ---------------------------------------------------------------------------
// LayerNorm (row length D): y = (x-mean)*rsqrt(var+eps)*g + b
// ---------------------------------------------------------------------------
__global__ __launch_bounds__(256)
void layernorm_kernel(const float* __restrict__ X, const float* __restrict__ g,
                      const float* __restrict__ beta, float* __restrict__ Y, int D)
{
    const int row = blockIdx.x;
    const float* x = X + (long long)row * D;
    __shared__ float s1[256], s2[256];
    const int tid = threadIdx.x;
    float a = 0.0f, b2 = 0.0f;
    for (int d = tid; d < D; d += 256) { float v = x[d]; a += v; b2 += v * v; }
    s1[tid] = a; s2[tid] = b2; __syncthreads();
    for (int st = 128; st > 0; st >>= 1) {
        if (tid < st) { s1[tid] += s1[tid + st]; s2[tid] += s2[tid + st]; }
        __syncthreads();
    }
    const float mean = s1[0] / (float)D;
    const float var  = s2[0] / (float)D - mean * mean;
    const float inv  = rsqrtf(var + 1e-5f);
    for (int d = tid; d < D; d += 256)
        Y[(long long)row * D + d] = (x[d] - mean) * inv * g[d] + beta[d];
}

// ---------------------------------------------------------------------------
// Causal softmax over rows of the [B*H*S, S] score matrix (in place)
// ---------------------------------------------------------------------------
__global__ __launch_bounds__(256)
void softmax_causal_kernel(float* __restrict__ P, int S)
{
    const long long row = blockIdx.x;
    const int q = (int)(row % S);
    const int L = q + 1;
    float* p = P + row * (long long)S;
    __shared__ float red[256];
    const int tid = threadIdx.x;

    float m = -3.4e38f;
    for (int j = tid; j < L; j += 256) m = fmaxf(m, p[j]);
    red[tid] = m; __syncthreads();
    for (int st = 128; st > 0; st >>= 1) {
        if (tid < st) red[tid] = fmaxf(red[tid], red[tid + st]);
        __syncthreads();
    }
    m = red[0]; __syncthreads();

    float s = 0.0f;
    for (int j = tid; j < L; j += 256) { float e = expf(p[j] - m); p[j] = e; s += e; }
    red[tid] = s; __syncthreads();
    for (int st = 128; st > 0; st >>= 1) {
        if (tid < st) red[tid] += red[tid + st];
        __syncthreads();
    }
    const float inv = 1.0f / red[0];
    for (int j = tid; j < L; j += 256) p[j] *= inv;
    for (int j = L + tid; j < S; j += 256) p[j] = 0.0f;
}

// ---------------------------------------------------------------------------
// Gate softmax + top-2 (ties -> lowest index, matching jax top_k)
// ---------------------------------------------------------------------------
__global__ __launch_bounds__(256)
void gate_top2_kernel(const float* __restrict__ X, const float* __restrict__ Wg,
                      int* __restrict__ idx, float* __restrict__ gv, int D, int E)
{
    const int t = blockIdx.x;
    const float* x = X + (long long)t * D;
    __shared__ float red[256 * 8];
    const int tid = threadIdx.x;
    float p[8];
    for (int e = 0; e < 8; ++e) p[e] = 0.0f;
    for (int d = tid; d < D; d += 256) {
        float xv = x[d];
        for (int e = 0; e < 8; ++e) p[e] += xv * Wg[d * E + e];
    }
    for (int e = 0; e < 8; ++e) red[tid * 8 + e] = p[e];
    __syncthreads();
    for (int st = 128; st > 0; st >>= 1) {
        if (tid < st)
            for (int e = 0; e < 8; ++e) red[tid * 8 + e] += red[(tid + st) * 8 + e];
        __syncthreads();
    }
    if (tid == 0) {
        float pr[8], mx = red[0];
        for (int e = 1; e < 8; ++e) mx = fmaxf(mx, red[e]);
        float s = 0.0f;
        for (int e = 0; e < 8; ++e) { pr[e] = expf(red[e] - mx); s += pr[e]; }
        for (int e = 0; e < 8; ++e) pr[e] /= s;
        int i0 = 0;
        for (int e = 1; e < 8; ++e) if (pr[e] > pr[i0]) i0 = e;
        int i1 = -1;
        for (int e = 0; e < 8; ++e) if (e != i0 && (i1 < 0 || pr[e] > pr[i1])) i1 = e;
        idx[t * 2]     = i0;  idx[t * 2 + 1] = i1;
        gv[t * 2]      = pr[i0]; gv[t * 2 + 1] = pr[i1];
    }
}

// ---------------------------------------------------------------------------
// Serial per-batch slot assignment (faithful to the reference cumsum semantics:
// choice-1 positions offset by TOTAL choice-0 count per expert; pos 1-based;
// keep iff pos < CAP; slot index = pos; slot 0 never used)
// ---------------------------------------------------------------------------
__global__ void assign_kernel(const int* __restrict__ idx, int* __restrict__ slotof,
                              int* __restrict__ slot2tok, int S, int E, int CAP)
{
    if (threadIdx.x != 0) return;
    const int b = blockIdx.x;
    int cnt[8], c1[8], cnt2[8];
    for (int e = 0; e < 8; ++e) { cnt[e] = 0; cnt2[e] = 0; }
    for (int s = 0; s < S; ++s) {
        const int t = b * S + s;
        const int e = idx[t * 2];
        const int p = ++cnt[e];
        if (p < CAP) { slotof[t * 2] = e * CAP + p; slot2tok[(b * E + e) * CAP + p] = s; }
        else         { slotof[t * 2] = -1; }
    }
    for (int e = 0; e < 8; ++e) c1[e] = cnt[e];
    for (int s = 0; s < S; ++s) {
        const int t = b * S + s;
        const int e = idx[t * 2 + 1];
        const int p = ++cnt2[e] + c1[e];
        if (p < CAP) { slotof[t * 2 + 1] = e * CAP + p; slot2tok[(b * E + e) * CAP + p] = s; }
        else         { slotof[t * 2 + 1] = -1; }
    }
}

__global__ __launch_bounds__(256)
void fill_i32_kernel(int* __restrict__ p, int n, int v)
{
    const int i = blockIdx.x * 256 + threadIdx.x;
    if (i < n) p[i] = v;
}

// ---------------------------------------------------------------------------
// Gather expert inputs (zero for empty slots) / scatter-combine expert outputs
// ---------------------------------------------------------------------------
__global__ __launch_bounds__(256)
void gather_kernel(const float* __restrict__ X1, const int* __restrict__ slot2tok,
                   float* __restrict__ EI, int S, int D, int E, int CAP)
{
    const int row = blockIdx.x;               // (b*E + e)*CAP + slot
    const int b   = row / (E * CAP);
    const int tok = slot2tok[row];
    float* dst = EI + (long long)row * D;
    if (tok < 0) {
        for (int d = threadIdx.x; d < D; d += 256) dst[d] = 0.0f;
    } else {
        const float* src = X1 + (long long)(b * S + tok) * D;
        for (int d = threadIdx.x; d < D; d += 256) dst[d] = src[d];
    }
}

__global__ __launch_bounds__(256)
void moe_combine_kernel(const float* __restrict__ X1, const float* __restrict__ EO,
                        const int* __restrict__ slotof, const float* __restrict__ gv,
                        float* __restrict__ OUT, int S, int D, int E, int CAP)
{
    const int t = blockIdx.x;
    const int b = t / S;
    const int s0 = slotof[t * 2], s1 = slotof[t * 2 + 1];
    const float g0 = gv[t * 2],  g1 = gv[t * 2 + 1];
    const float* e0 = (s0 >= 0) ? EO + ((long long)b * E * CAP + s0) * D : nullptr;
    const float* e1 = (s1 >= 0) ? EO + ((long long)b * E * CAP + s1) * D : nullptr;
    for (int d = threadIdx.x; d < D; d += 256) {
        float v = X1[(long long)t * D + d];
        if (e0) v += g0 * e0[d];
        if (e1) v += g1 * e1[d];
        OUT[(long long)t * D + d] = v;
    }
}

// ---------------------------------------------------------------------------
// Per-token NLL (log-softmax over V) and final sum
// ---------------------------------------------------------------------------
__global__ __launch_bounds__(256)
void nll_kernel(const float* __restrict__ logits, const int* __restrict__ y,
                float* __restrict__ rowloss, int V)
{
    const int t = blockIdx.x;
    const float* lg = logits + (long long)t * V;
    __shared__ float red[256];
    const int tid = threadIdx.x;
    float m = -3.4e38f;
    for (int j = tid; j < V; j += 256) m = fmaxf(m, lg[j]);
    red[tid] = m; __syncthreads();
    for (int st = 128; st > 0; st >>= 1) {
        if (tid < st) red[tid] = fmaxf(red[tid], red[tid + st]);
        __syncthreads();
    }
    m = red[0]; __syncthreads();
    float s = 0.0f;
    for (int j = tid; j < V; j += 256) s += expf(lg[j] - m);
    red[tid] = s; __syncthreads();
    for (int st = 128; st > 0; st >>= 1) {
        if (tid < st) red[tid] += red[tid + st];
        __syncthreads();
    }
    if (tid == 0) rowloss[t] = logf(red[0]) + m - lg[y[t]];
}

__global__ __launch_bounds__(256)
void sum_kernel(const float* __restrict__ r, float* __restrict__ out, int n)
{
    __shared__ float s[256];
    const int tid = threadIdx.x;
    float a = 0.0f;
    for (int i = tid; i < n; i += 256) a += r[i];
    s[tid] = a; __syncthreads();
    for (int st = 128; st > 0; st >>= 1) {
        if (tid < st) s[tid] += s[tid + st];
        __syncthreads();
    }
    if (tid == 0) out[0] = s[0];
}

// ---------------------------------------------------------------------------
// Host orchestration
// ---------------------------------------------------------------------------
static inline void launch_gemm(hipStream_t st,
    const float* A, int lda, long long sA1, long long sA2,
    const float* B, int ldb, long long sB1, long long sB2, int transB,
    float* C, int ldc, long long sC1, long long sC2,
    const float* bias, const float* resid, int ldr, long long sR1, long long sR2,
    int M, int N, int K, int batch, int batchDiv, float alpha, int relu)
{
    dim3 grid(N / 64, M / 128, batch), blk(128);
    hipLaunchKernelGGL(wmma_gemm_kernel, grid, blk, 0, st,
        A, lda, sA1, sA2, B, ldb, sB1, sB2, transB,
        C, ldc, sC1, sC2, bias, resid, ldr, sR1, sR2,
        M, N, K, batchDiv, alpha, relu);
}

extern "C" void kernel_launch(void* const* d_in, const int* in_sizes, int n_in,
                              void* d_out, int out_size, void* d_ws, size_t ws_size,
                              hipStream_t stream)
{
    (void)in_sizes; (void)n_in; (void)out_size; (void)ws_size;
    constexpr int Bc = 2, Sc = 1024, Dc = 1024, Hc = 16, DHc = 64;
    constexpr int HIDc = 4096, Ec = 8, CAPc = 384, Vc = 32000, NLc = 4;
    constexpr int BS = Bc * Sc;                   // 2048 token rows

    // ---- unpack inputs (setup_inputs dict order, nested insertion order) ----
    int ci = 0;
    const int*   x   = (const int*)d_in[ci++];
    const int*   y   = (const int*)d_in[ci++];
    const float* emb = (const float*)d_in[ci++];
    struct LayerP {
        const float *wq,*wk,*wv,*bq,*bk,*bv,*wo,*bo,*ln1g,*ln1b,*ln2g,*ln2b;
        const float *fw1,*fb1,*fw2,*fb2,*gate,*ew1,*ew2;
    } L[NLc];
    for (int i = 0; i < NLc; ++i) {
        L[i].wq   = (const float*)d_in[ci++]; L[i].wk   = (const float*)d_in[ci++];
        L[i].wv   = (const float*)d_in[ci++]; L[i].bq   = (const float*)d_in[ci++];
        L[i].bk   = (const float*)d_in[ci++]; L[i].bv   = (const float*)d_in[ci++];
        L[i].wo   = (const float*)d_in[ci++]; L[i].bo   = (const float*)d_in[ci++];
        L[i].ln1g = (const float*)d_in[ci++]; L[i].ln1b = (const float*)d_in[ci++];
        L[i].ln2g = (const float*)d_in[ci++]; L[i].ln2b = (const float*)d_in[ci++];
        L[i].fw1 = L[i].fb1 = L[i].fw2 = L[i].fb2 = L[i].gate = L[i].ew1 = L[i].ew2 = nullptr;
        if (i % 2 == 0) {
            L[i].fw1 = (const float*)d_in[ci++]; L[i].fb1 = (const float*)d_in[ci++];
            L[i].fw2 = (const float*)d_in[ci++]; L[i].fb2 = (const float*)d_in[ci++];
        } else {
            L[i].gate = (const float*)d_in[ci++];
            L[i].ew1  = (const float*)d_in[ci++];
            L[i].ew2  = (const float*)d_in[ci++];
        }
    }
    const float* dec_w = (const float*)d_in[ci++];
    const float* dec_b = (const float*)d_in[ci++];

    // ---- workspace layout ----
    char* base = (char*)d_ws;
    size_t off = 0;
    auto alloc = [&](size_t bytes) -> char* {
        char* p = base + off;
        off += (bytes + 255) & ~(size_t)255;
        return p;
    };
    float* Hb  = (float*)alloc((size_t)BS * Dc * 4);                 // activations
    float* X1  = (float*)alloc((size_t)BS * Dc * 4);                 // post-LN1
    float* Qb  = (float*)alloc((size_t)BS * Dc * 4);
    float* Kb  = (float*)alloc((size_t)BS * Dc * 4);
    float* Vb  = (float*)alloc((size_t)BS * Dc * 4);
    float* AOb = (float*)alloc((size_t)BS * Dc * 4);                 // attn ctx
    float* FF  = (float*)alloc((size_t)BS * HIDc * 4);               // ffn hidden / expert inputs
    float* FF2 = (float*)alloc((size_t)Bc * Ec * CAPc * Dc * 4);     // expert outputs
    float* BIG = (float*)alloc((size_t)BS * Vc * 4);                 // logits / scores / expert hidden
    int*   top2i    = (int*)  alloc((size_t)BS * 2 * 4);
    float* top2g    = (float*)alloc((size_t)BS * 2 * 4);
    int*   slotof   = (int*)  alloc((size_t)BS * 2 * 4);
    int*   slot2tok = (int*)  alloc((size_t)Bc * Ec * CAPc * 4);
    float* rowloss  = (float*)alloc((size_t)BS * 4);

    // ---- embedding ----
    hipLaunchKernelGGL(embed_kernel, dim3(BS), dim3(256), 0, stream, x, emb, Hb, Sc, Dc);

    // ---- layers ----
    for (int i = 0; i < NLc; ++i) {
        const LayerP& P = L[i];

        // Q,K,V projections: [2048,1024] x [1024,1024] + bias
        launch_gemm(stream, Hb, Dc, 0, 0, P.wq, Dc, 0, 0, 0, Qb, Dc, 0, 0,
                    P.bq, nullptr, 0, 0, 0, BS, Dc, Dc, 1, 1, 1.0f, 0);
        launch_gemm(stream, Hb, Dc, 0, 0, P.wk, Dc, 0, 0, 0, Kb, Dc, 0, 0,
                    P.bk, nullptr, 0, 0, 0, BS, Dc, Dc, 1, 1, 1.0f, 0);
        launch_gemm(stream, Hb, Dc, 0, 0, P.wv, Dc, 0, 0, 0, Vb, Dc, 0, 0,
                    P.bv, nullptr, 0, 0, 0, BS, Dc, Dc, 1, 1, 1.0f, 0);

        // scores[b,h] = Q_bh (SxDH) * K_bh^T / sqrt(DH);  batch z = b*H + h
        launch_gemm(stream,
                    Qb, Dc, (long long)Sc * Dc, DHc,
                    Kb, Dc, (long long)Sc * Dc, DHc, /*transB=*/1,
                    BIG, Sc, (long long)Hc * Sc * Sc, (long long)Sc * Sc,
                    nullptr, nullptr, 0, 0, 0,
                    Sc, Sc, DHc, Bc * Hc, Hc, 0.125f, 0);

        // causal softmax over B*H*S rows
        hipLaunchKernelGGL(softmax_causal_kernel, dim3(Bc * Hc * Sc), dim3(256), 0, stream,
                           BIG, Sc);

        // ctx[b,h] = probs (SxS) * V_bh (SxDH)
        launch_gemm(stream,
                    BIG, Sc, (long long)Hc * Sc * Sc, (long long)Sc * Sc,
                    Vb, Dc, (long long)Sc * Dc, DHc, 0,
                    AOb, Dc, (long long)Sc * Dc, DHc,
                    nullptr, nullptr, 0, 0, 0,
                    Sc, DHc, Sc, Bc * Hc, Hc, 1.0f, 0);

        // output projection + bias + residual(Hb) -> Qb ; then LN1 -> X1
        launch_gemm(stream, AOb, Dc, 0, 0, P.wo, Dc, 0, 0, 0, Qb, Dc, 0, 0,
                    P.bo, Hb, Dc, 0, 0, BS, Dc, Dc, 1, 1, 1.0f, 0);
        hipLaunchKernelGGL(layernorm_kernel, dim3(BS), dim3(256), 0, stream,
                           Qb, P.ln1g, P.ln1b, X1, Dc);

        if (i % 2 == 0) {
            // dense FFN: relu(X1@fw1 + fb1) @ fw2 + fb2 + X1 ; LN2 -> Hb
            launch_gemm(stream, X1, Dc, 0, 0, P.fw1, HIDc, 0, 0, 0, FF, HIDc, 0, 0,
                        P.fb1, nullptr, 0, 0, 0, BS, HIDc, Dc, 1, 1, 1.0f, 1);
            launch_gemm(stream, FF, HIDc, 0, 0, P.fw2, Dc, 0, 0, 0, Qb, Dc, 0, 0,
                        P.fb2, X1, Dc, 0, 0, BS, Dc, HIDc, 1, 1, 1.0f, 0);
            hipLaunchKernelGGL(layernorm_kernel, dim3(BS), dim3(256), 0, stream,
                               Qb, P.ln2g, P.ln2b, Hb, Dc);
        } else {
            // MoE: gating, capacity assignment, gather, expert FFN, combine
            hipLaunchKernelGGL(gate_top2_kernel, dim3(BS), dim3(256), 0, stream,
                               X1, P.gate, top2i, top2g, Dc, Ec);
            const int nslots = Bc * Ec * CAPc;
            hipLaunchKernelGGL(fill_i32_kernel, dim3((nslots + 255) / 256), dim3(256), 0, stream,
                               slot2tok, nslots, -1);
            hipLaunchKernelGGL(assign_kernel, dim3(Bc), dim3(32), 0, stream,
                               top2i, slotof, slot2tok, Sc, Ec, CAPc);
            hipLaunchKernelGGL(gather_kernel, dim3(nslots), dim3(256), 0, stream,
                               X1, slot2tok, FF, Sc, Dc, Ec, CAPc);

            // expert GEMM1: relu(EI[b,e] (CAPxD) @ ew1[e] (DxHID)) -> BIG ; z=b*E+e
            launch_gemm(stream,
                        FF, Dc, (long long)Ec * CAPc * Dc, (long long)CAPc * Dc,
                        P.ew1, HIDc, 0, (long long)Dc * HIDc, 0,
                        BIG, HIDc, (long long)Ec * CAPc * HIDc, (long long)CAPc * HIDc,
                        nullptr, nullptr, 0, 0, 0,
                        CAPc, HIDc, Dc, Bc * Ec, Ec, 1.0f, 1);
            // expert GEMM2: EH (CAPxHID) @ ew2[e] (HIDxD) -> FF2
            launch_gemm(stream,
                        BIG, HIDc, (long long)Ec * CAPc * HIDc, (long long)CAPc * HIDc,
                        P.ew2, Dc, 0, (long long)HIDc * Dc, 0,
                        FF2, Dc, (long long)Ec * CAPc * Dc, (long long)CAPc * Dc,
                        nullptr, nullptr, 0, 0, 0,
                        CAPc, Dc, HIDc, Bc * Ec, Ec, 1.0f, 0);

            hipLaunchKernelGGL(moe_combine_kernel, dim3(BS), dim3(256), 0, stream,
                               X1, FF2, slotof, top2g, Qb, Sc, Dc, Ec, CAPc);
            hipLaunchKernelGGL(layernorm_kernel, dim3(BS), dim3(256), 0, stream,
                               Qb, P.ln2g, P.ln2b, Hb, Dc);
        }
    }

    // ---- decoder + loss ----
    launch_gemm(stream, Hb, Dc, 0, 0, dec_w, Vc, 0, 0, 0, BIG, Vc, 0, 0,
                dec_b, nullptr, 0, 0, 0, BS, Vc, Dc, 1, 1, 1.0f, 0);
    hipLaunchKernelGGL(nll_kernel, dim3(BS), dim3(256), 0, stream, BIG, y, rowloss, Vc);
    hipLaunchKernelGGL(sum_kernel, dim3(1), dim3(256), 0, stream, rowloss, (float*)d_out, BS);
}
